// MetaStateStep_71347996721420
// MI455X (gfx1250) — matrix operand order repaired and checked by reading, not compiled, not run on hardware
//
#include <hip/hip_runtime.h>
#include <hip/hip_bf16.h>

// ---------------------------------------------------------------------------
// MI455X (gfx1250). All eight matmuls run through one templated bf16 WMMA
// GEMM (v_wmma_f32_16x16x32_bf16, f32 accumulate), 128x64 block tile,
// 32x32 per wave (4 WMMA / 4 frag-loads per K-chunk). Global->LDS staging
// uses GLOBAL_LOAD_ASYNC_TO_LDS_B128 (ASYNCcnt / s_wait_asynccnt).
// ---------------------------------------------------------------------------

typedef __attribute__((ext_vector_type(16))) __bf16 bf16x16;
typedef __attribute__((ext_vector_type(8)))  float  f32x8;

union ABFrag { bf16x16 v; unsigned int u[8]; };

#define BM 128
#define BN 64
#define BK 32
#define SA  40   // LDS row stride (elements) for A tile
#define SBT 40   // LDS row stride for B tile, N-major (W stored N x K)
#define SBK 72   // LDS row stride for B tile, K-major (B stored K x N)

enum { EPI_BF16 = 0, EPI_RESID = 1, EPI_GELU = 2, EPI_DECAY = 3 };
enum { B_TRANS = 0, B_KMAJOR = 1 };

// Async 16B global -> LDS copy (per lane). LDS byte address = low 32 bits of
// the flat pointer (ISA: LDS aperture truncates addr[31:0]).
__device__ __forceinline__ void async_ld16(unsigned lds_off, const void* gptr) {
  asm volatile("global_load_async_to_lds_b128 %0, %1, off"
               :: "v"(lds_off), "v"(gptr)
               : "memory");
}
__device__ __forceinline__ void wait_async0() {
  asm volatile("s_wait_asynccnt 0x0" ::: "memory");
}

// C/D 16x16 f32 layout: lane l, vgpr r -> M = r + 8*(l>>4), N = l&15
// A 16x32 bf16 layout : lane l, vgpr v, half b ->
//   M = l&15,  K = ((l>>4)&1)*16 + v*2 + b
// B 32x16 bf16 layout : lane l, vgpr v, half b ->
//   N = l&15,  K = (l>>4)*16 + v*2 + b

template <int EPI, int BMODE>
__global__ __launch_bounds__(256) void gemm_wmma_bf16(
    const __bf16* __restrict__ A,   // M x K, row-major
    const __bf16* __restrict__ Bm,  // B_TRANS: N x K row-major ; B_KMAJOR: K x N row-major
    int M, int N, int K,
    float* __restrict__ outF, __bf16* __restrict__ outB,
    const float* __restrict__ resid, const float* __restrict__ bias,
    const float* __restrict__ s0, const float* __restrict__ s1,
    const float* __restrict__ dlogit) {
  __shared__ __align__(16) unsigned short As[BM * SA];
  __shared__ __align__(16) unsigned short Bs[(BMODE == B_TRANS) ? (BN * SBT) : (BK * SBK)];

  const int tid  = threadIdx.x;
  const int lane = tid & 31;
  const int wave = tid >> 5;
  const int wr   = wave >> 1;  // 0..3 : 32-row slice
  const int wc   = wave & 1;   // 0..1 : 32-col slice
  const int bm0  = blockIdx.y * BM;
  const int bn0  = blockIdx.x * BN;

  f32x8 acc[2][2] = {};

  const int ar  = tid >> 2;        // 0..63
  const int ac  = (tid & 3) << 3;  // 0,8,16,24
  const int bkr = tid >> 3;        // 0..31
  const int bkc = (tid & 7) << 3;  // 0..56

  const unsigned asbase = (unsigned)(unsigned long long)&As[0];
  const unsigned bsbase = (unsigned)(unsigned long long)&Bs[0];

  for (int k0 = 0; k0 < K; k0 += BK) {
    __syncthreads();  // previous tile fully consumed
    // A tile: 128x32, two 16B async transfers per thread
    async_ld16(asbase + (unsigned)((ar * SA + ac) * 2),
               A + (size_t)(bm0 + ar) * K + k0 + ac);
    async_ld16(asbase + (unsigned)(((ar + 64) * SA + ac) * 2),
               A + (size_t)(bm0 + ar + 64) * K + k0 + ac);
    if (BMODE == B_TRANS) {
      async_ld16(bsbase + (unsigned)((ar * SBT + ac) * 2),
                 Bm + (size_t)(bn0 + ar) * K + k0 + ac);
    } else {
      async_ld16(bsbase + (unsigned)((bkr * SBK + bkc) * 2),
                 Bm + (size_t)(k0 + bkr) * N + bn0 + bkc);
    }
    if (k0 + BK < K) {
      __builtin_prefetch(A + (size_t)(bm0 + ar) * K + k0 + BK + ac, 0, 1);
    }
    wait_async0();    // this wave's async copies landed in LDS
    __syncthreads();  // everyone's copies landed

    // two A fragments (rows wr*32 .. wr*32+31)
    ABFrag af[2];
#pragma unroll
    for (int rt = 0; rt < 2; ++rt) {
      const int am = wr * 32 + rt * 16 + (lane & 15);
#pragma unroll
      for (int v = 0; v < 8; ++v) {
        const int koff = (((lane >> 4) & 1) << 4) | (v << 1);
        af[rt].u[v] = *(const unsigned int*)&As[am * SA + koff];
      }
    }
    // two B fragments (cols wc*32 .. wc*32+31)
    ABFrag bf_[2];
#pragma unroll
    for (int nt = 0; nt < 2; ++nt) {
      const int bn = wc * 32 + nt * 16 + (lane & 15);
      if (BMODE == B_TRANS) {
#pragma unroll
        for (int v = 0; v < 8; ++v) {
          const int koff = (((lane >> 4) & 1) << 4) | (v << 1);
          bf_[nt].u[v] = *(const unsigned int*)&Bs[bn * SBT + koff];
        }
      } else {
#pragma unroll
        for (int v = 0; v < 8; ++v) {
          const int kk = (((lane >> 4) & 1) << 4) | (v << 1);
          const unsigned int lo = Bs[kk * SBK + bn];
          const unsigned int hi = Bs[(kk + 1) * SBK + bn];
          bf_[nt].u[v] = lo | (hi << 16);
        }
      }
    }
    // 4 WMMAs per K-chunk
#pragma unroll
    for (int rt = 0; rt < 2; ++rt)
#pragma unroll
      for (int nt = 0; nt < 2; ++nt)
        acc[rt][nt] = __builtin_amdgcn_wmma_f32_16x16x32_bf16(
            false, af[rt].v, false, bf_[nt].v, (short)0, acc[rt][nt], false, false);
  }

  // ---- fused epilogues ----
  float alpha = 1.0f;
  if (EPI == EPI_RESID) alpha = s0[0] * s1[0];
  float logdecay = 0.0f;
  if (EPI == EPI_DECAY) {
    const float d = 1.0f / (1.0f + __expf(-dlogit[0]));  // sigmoid
    logdecay = __logf(d);
  }

#pragma unroll
  for (int rt = 0; rt < 2; ++rt) {
#pragma unroll
    for (int nt = 0; nt < 2; ++nt) {
      const f32x8 acct = acc[rt][nt];
      const int n = bn0 + wc * 32 + nt * 16 + (lane & 15);
#pragma unroll
      for (int r = 0; r < 8; ++r) {
        const int m = bm0 + wr * 32 + rt * 16 + ((lane >> 4) & 1) * 8 + r;
        const size_t idx = (size_t)m * N + n;
        const float f = acct[r];
        if (EPI == EPI_BF16) {
          outB[idx] = (__bf16)f;
        } else if (EPI == EPI_RESID) {
          outF[idx] = resid[idx] + alpha * f;
        } else if (EPI == EPI_GELU) {
          const float h = f + bias[n];
          const float g = 0.5f * h * (1.0f + erff(h * 0.70710678118654752f));
          outB[idx] = (__bf16)g;
        } else {  // EPI_DECAY: m = t, n = s (per-batch launch)
          const float w = (n > m) ? __expf((float)(n - m - 1) * logdecay) : 0.0f;
          outB[idx] = (__bf16)(f * w);
        }
      }
    }
  }
}

// Row-wise RMS norm over the last dim (V = 2048), f32 in -> bf16 out.
__global__ __launch_bounds__(256) void rmsnorm_to_bf16(
    const float* __restrict__ x, __bf16* __restrict__ out, int Vdim) {
  const int row = blockIdx.x;
  const float* xr = x + (size_t)row * Vdim;
  __bf16* orow = out + (size_t)row * Vdim;
  const int tid = threadIdx.x;

  const float4 a = ((const float4*)xr)[tid * 2 + 0];
  const float4 b = ((const float4*)xr)[tid * 2 + 1];
  float ss = a.x * a.x + a.y * a.y + a.z * a.z + a.w * a.w +
             b.x * b.x + b.y * b.y + b.z * b.z + b.w * b.w;

#pragma unroll
  for (int off = 16; off >= 1; off >>= 1) ss += __shfl_down(ss, off, 32);

  __shared__ float red[8];
  if ((tid & 31) == 0) red[tid >> 5] = ss;
  __syncthreads();
  if (tid == 0) {
    float t = 0.0f;
#pragma unroll
    for (int i = 0; i < 8; ++i) t += red[i];
    red[0] = t;
  }
  __syncthreads();
  const float rinv = rsqrtf(red[0] / (float)Vdim + 1.1920929e-07f);

  union { uint4 q; __bf16 h[8]; } pk;
  pk.h[0] = (__bf16)(a.x * rinv); pk.h[1] = (__bf16)(a.y * rinv);
  pk.h[2] = (__bf16)(a.z * rinv); pk.h[3] = (__bf16)(a.w * rinv);
  pk.h[4] = (__bf16)(b.x * rinv); pk.h[5] = (__bf16)(b.y * rinv);
  pk.h[6] = (__bf16)(b.z * rinv); pk.h[7] = (__bf16)(b.w * rinv);
  ((uint4*)orow)[tid] = pk.q;
}

__global__ __launch_bounds__(256) void f32_to_bf16(
    const float* __restrict__ in, __bf16* __restrict__ out, int n) {
  const int i = blockIdx.x * blockDim.x + threadIdx.x;
  if (i < n) out[i] = (__bf16)in[i];
}

// ---------------------------------------------------------------------------
extern "C" void kernel_launch(void* const* d_in, const int* in_sizes, int n_in,
                              void* d_out, int out_size, void* d_ws, size_t ws_size,
                              hipStream_t stream) {
  (void)in_sizes; (void)n_in; (void)out_size; (void)ws_size;
  constexpr int Bb = 4, T = 2048, V = 2048, D = 256, R = 512;
  const int M = Bb * T;  // 8192

  const float* x       = (const float*)d_in[0];
  const float* Wq      = (const float*)d_in[1];
  const float* Wk      = (const float*)d_in[2];
  const float* Wv      = (const float*)d_in[3];
  const float* Wo      = (const float*)d_in[4];
  const float* dlogit  = (const float*)d_in[5];
  const float* q_os    = (const float*)d_in[6];
  const float* Wdown   = (const float*)d_in[7];
  const float* Wup     = (const float*)d_in[8];
  const float* t_bias  = (const float*)d_in[9];
  const float* t_os    = (const float*)d_in[10];
  const float* q_scale = (const float*)d_in[11];
  const float* t_scale = (const float*)d_in[12];
  float* out = (float*)d_out;

  // Workspace layout (bytes); total ~88 MB.
  char* ws = (char*)d_ws;
  size_t o = 0;
  __bf16* Wq_b  = (__bf16*)(ws + o); o += (size_t)D * V * 2;  // 1 MB
  __bf16* Wk_b  = (__bf16*)(ws + o); o += (size_t)D * V * 2;  // 1 MB
  __bf16* Wv_b  = (__bf16*)(ws + o); o += (size_t)D * V * 2;  // 1 MB
  __bf16* Wo_b  = (__bf16*)(ws + o); o += (size_t)V * D * 2;  // 1 MB
  __bf16* Wdn_b = (__bf16*)(ws + o); o += (size_t)R * V * 2;  // 2 MB
  __bf16* Wup_b = (__bf16*)(ws + o); o += (size_t)V * R * 2;  // 2 MB
  __bf16* xn_b  = (__bf16*)(ws + o); o += (size_t)M * V * 2;  // 32 MB (reused for 2nd norm)
  __bf16* q_b   = (__bf16*)(ws + o); o += (size_t)M * D * 2;  // 4 MB
  __bf16* k_b   = (__bf16*)(ws + o); o += (size_t)M * D * 2;  // 4 MB
  __bf16* v_b   = (__bf16*)(ws + o); o += (size_t)M * D * 2;  // 4 MB
  __bf16* ret_b = (__bf16*)(ws + o); o += (size_t)M * D * 2;  // 4 MB
  __bf16* sc_b  = (__bf16*)(ws + o); o += (size_t)Bb * T * T * 2;  // 32 MB
  __bf16* h_b   = sc_b;  // reuse: scores dead before h is produced (8 MB needed)

  // 1) weights f32 -> bf16
  f32_to_bf16<<<(D * V + 255) / 256, 256, 0, stream>>>(Wq, Wq_b, D * V);
  f32_to_bf16<<<(D * V + 255) / 256, 256, 0, stream>>>(Wk, Wk_b, D * V);
  f32_to_bf16<<<(D * V + 255) / 256, 256, 0, stream>>>(Wv, Wv_b, D * V);
  f32_to_bf16<<<(V * D + 255) / 256, 256, 0, stream>>>(Wo, Wo_b, V * D);
  f32_to_bf16<<<(R * V + 255) / 256, 256, 0, stream>>>(Wdown, Wdn_b, R * V);
  f32_to_bf16<<<(V * R + 255) / 256, 256, 0, stream>>>(Wup, Wup_b, V * R);

  // 2) xn = rms_norm(x)
  rmsnorm_to_bf16<<<M, 256, 0, stream>>>(x, xn_b, V);

  // 3) q/k/v = xn @ W^T   (M=8192, N=256, K=2048)
  gemm_wmma_bf16<EPI_BF16, B_TRANS><<<dim3(D / BN, M / BM), 256, 0, stream>>>(
      xn_b, Wq_b, M, D, V, nullptr, q_b, nullptr, nullptr, nullptr, nullptr, nullptr);
  gemm_wmma_bf16<EPI_BF16, B_TRANS><<<dim3(D / BN, M / BM), 256, 0, stream>>>(
      xn_b, Wk_b, M, D, V, nullptr, k_b, nullptr, nullptr, nullptr, nullptr, nullptr);
  gemm_wmma_bf16<EPI_BF16, B_TRANS><<<dim3(D / BN, M / BM), 256, 0, stream>>>(
      xn_b, Wv_b, M, D, V, nullptr, v_b, nullptr, nullptr, nullptr, nullptr, nullptr);

  // 4) per-batch: scores = (q k^T) * decay mask ; retrieved = scores @ v
  for (int b = 0; b < Bb; ++b) {
    const size_t od = (size_t)b * T * D;
    const size_t os = (size_t)b * T * T;
    gemm_wmma_bf16<EPI_DECAY, B_TRANS><<<dim3(T / BN, T / BM), 256, 0, stream>>>(
        q_b + od, k_b + od, T, T, D, nullptr, sc_b + os,
        nullptr, nullptr, nullptr, nullptr, dlogit);
    gemm_wmma_bf16<EPI_BF16, B_KMAJOR><<<dim3(D / BN, T / BM), 256, 0, stream>>>(
        sc_b + os, v_b + od, T, D, T, nullptr, ret_b + od,
        nullptr, nullptr, nullptr, nullptr, nullptr);
  }

  // 5) x2 = x + q_scale*q_out_scale * (retrieved @ Wo^T)  -> d_out (f32)
  gemm_wmma_bf16<EPI_RESID, B_TRANS><<<dim3(V / BN, M / BM), 256, 0, stream>>>(
      ret_b, Wo_b, M, V, D, out, nullptr, x, nullptr, q_scale, q_os, nullptr);

  // 6) xn2 = rms_norm(x2)
  rmsnorm_to_bf16<<<M, 256, 0, stream>>>(out, xn_b, V);

  // 7) h = gelu(xn2 @ Wdown^T + bias)   (N=512, K=2048)
  gemm_wmma_bf16<EPI_GELU, B_TRANS><<<dim3(R / BN, M / BM), 256, 0, stream>>>(
      xn_b, Wdn_b, M, R, V, nullptr, h_b, nullptr, t_bias, nullptr, nullptr, nullptr);

  // 8) out = x2 + t_scale*t_out_scale * (h @ Wup^T)  (in-place residual on d_out)
  gemm_wmma_bf16<EPI_RESID, B_TRANS><<<dim3(V / BN, M / BM), 256, 0, stream>>>(
      h_b, Wup_b, M, V, R, out, nullptr, out, nullptr, t_scale, t_os, nullptr);
}
